// RandomBipartiteSoftMatching_30193620091390
// MI455X (gfx1250) — matrix operand order repaired and checked by reading, not compiled, not run on hardware
//
#include <hip/hip_runtime.h>
#include <hip/hip_bf16.h>

// Problem constants (match reference)
#define BB 16
#define NN 2048
#define CC 1024
#define RR 1024
#define NBT (NN - RR)   // 1024 destination rows

typedef __attribute__((ext_vector_type(16))) _Float16 v16h;
typedef __attribute__((ext_vector_type(8)))  float    v8f;
typedef __attribute__((ext_vector_type(4))) unsigned int u32x4;
typedef __attribute__((ext_vector_type(8))) int          i32x8;
typedef __attribute__((ext_vector_type(4))) int          i32x4;

#if __has_builtin(__builtin_amdgcn_tensor_load_to_lds) && \
    __has_builtin(__builtin_amdgcn_s_wait_tensorcnt)
#define USE_TDM 1
#else
#define USE_TDM 0
#endif

// ---------------------------------------------------------------------------
// K1: row-normalize metric, store as f16 (halves bandwidth for matrix stages)
// ---------------------------------------------------------------------------
__global__ __launch_bounds__(256) void k_normalize(const float* __restrict__ x,
                                                   _Float16* __restrict__ m) {
  const int row = blockIdx.x;                 // b*NN + n
  const float* xr = x + (size_t)row * CC;
  _Float16* mr = m + (size_t)row * CC;

  float ss = 0.f;
  for (int c = threadIdx.x; c < CC; c += 256) { float v = xr[c]; ss += v * v; }
  for (int off = 16; off > 0; off >>= 1) ss += __shfl_down(ss, off, 32);
  __shared__ float red[8];
  if ((threadIdx.x & 31) == 0) red[threadIdx.x >> 5] = ss;
  __syncthreads();
  float tot = 0.f;
#pragma unroll
  for (int i = 0; i < 8; ++i) tot += red[i];
  const float rn = rsqrtf(tot);
  for (int c = threadIdx.x; c < CC; c += 256)
    mr[c] = (_Float16)(xr[c] * rn);
}

// ---------------------------------------------------------------------------
// K2: S[b][c] = sum_n m[b][n][c]   (column sums of normalized rows)
// ---------------------------------------------------------------------------
__global__ __launch_bounds__(256) void k_colsum(const _Float16* __restrict__ m,
                                                float* __restrict__ S) {
  const int t = blockIdx.x * 256 + threadIdx.x;   // b*CC + c
  const int b = t / CC;
  const int c = t % CC;
  const _Float16* p = m + (size_t)b * NN * CC + c;
  float s = 0.f;
  for (int n = 0; n < NN; ++n) s += (float)p[(size_t)n * CC];
  S[t] = s;
}

// ---------------------------------------------------------------------------
// K3: key[b][n] = -(m[n]·S[b] - 1)   (negated => ascending sort == top score)
// One wave per row. (sim.sum(-1)-diag collapses to m·S - 1 since sim[n,n]=1.)
// ---------------------------------------------------------------------------
__global__ __launch_bounds__(256) void k_rowscore(const _Float16* __restrict__ m,
                                                  const float* __restrict__ S,
                                                  float* __restrict__ key) {
  const int wid  = (blockIdx.x * 256 + threadIdx.x) >> 5;  // global row id
  const int lane = threadIdx.x & 31;
  const int b = wid / NN;
  const int n = wid % NN;
  const _Float16* mr = m + ((size_t)b * NN + n) * CC;
  const float* Sb = S + (size_t)b * CC;
  float s = 0.f;
  for (int c = lane; c < CC; c += 32) s += (float)mr[c] * Sb[c];
  for (int off = 16; off > 0; off >>= 1) s += __shfl_down(s, off, 32);
  if (lane == 0) key[wid] = -(s - 1.0f);
}

// ---------------------------------------------------------------------------
// K4: per-batch bitonic argsort (ascending on negated key) in LDS.
// 1024 threads, 2048 elements, 16KB LDS.
// ---------------------------------------------------------------------------
__global__ __launch_bounds__(1024) void k_sort(const float* __restrict__ key,
                                               int* __restrict__ idx) {
  __shared__ float sk[NN];
  __shared__ int   sv[NN];
  const int b = blockIdx.x;
  const int t = threadIdx.x;
  for (int i = t; i < NN; i += 1024) { sk[i] = key[b * NN + i]; sv[i] = i; }
  __syncthreads();
  for (int size = 2; size <= NN; size <<= 1) {
    for (int stride = size >> 1; stride > 0; stride >>= 1) {
      const int pos = 2 * t - (t & (stride - 1));
      const bool up = ((pos & size) == 0);
      float k1 = sk[pos], k2 = sk[pos + stride];
      const bool sw = up ? (k1 > k2) : (k1 < k2);
      if (sw) {
        int v1 = sv[pos], v2 = sv[pos + stride];
        sk[pos] = k2; sk[pos + stride] = k1;
        sv[pos] = v2; sv[pos + stride] = v1;
      }
      __syncthreads();
    }
  }
  for (int i = t; i < NN; i += 1024) idx[b * NN + i] = sv[i];
}

// ---------------------------------------------------------------------------
// K5: scores = a @ b^T per batch via V_WMMA_F32_16X16X32_F16, fused row argmax.
//   8 waves/block; wave w owns A rows [mblk*128 + w*16, +16) gathered by a_idx.
//   B tiles (16 gathered rows x CC f16 = 32KB) are staged into LDS by the
//   Tensor Data Mover in GATHER MODE with DOUBLE BUFFERING: wave 0 issues the
//   tensor_load_to_lds for tile j+1, then s_wait_tensorcnt(1) (in-order
//   TENSORcnt completion => tile j resident) so the DMA for j+1 overlaps the
//   32-WMMA compute phase. An empty volatile asm escapes the LDS pointer so
//   the optimizer treats the TDM op as a writer of Bt (otherwise the B-tile
//   reads are "never stored" undef and get hoisted out of the loop).
// ---------------------------------------------------------------------------
#define TILE_BYTES (16 * CC * 2)   // one B tile in LDS

#if USE_TDM
__device__ __forceinline__ void tdm_issue_tile(const _Float16* mb,
                                               const int* b_idx, int jj) {
  // Row indices of tile jj, loaded uniformly (land in SGPRs).
  unsigned ridx[16];
#pragma unroll
  for (int k = 0; k < 16; ++k) ridx[k] = (unsigned)b_idx[jj * 16 + k];

  // ---- D# group 0: count=1 | gather_mode=1 (16-bit idx);
  //      lds_addr = buffer parity; global_addr = batch base; type=2 ("image")
  const unsigned long long ga = (unsigned long long)(uintptr_t)mb;
  u32x4 g0;
  g0[0] = 0x80000001u;
  g0[1] = (unsigned)((jj & 1) * TILE_BYTES);
  g0[2] = (unsigned)(ga & 0xffffffffu);
  g0[3] = (unsigned)((ga >> 32) & 0x01ffffffu) | (2u << 30);

  // ---- D# group 1: data_size=2B; tensor_dim0=CC; tensor_dim1=NN;
  //      tile_dim0=CC; tile_dim1=16 (#indices); dim0_stride=CC
  i32x8 g1;
  g1[0] = 1 << 16;           // data_size = 2 bytes
  g1[1] = CC << 16;          // tensor_dim0[15:0]
  g1[2] = NN << 16;          // tensor_dim0[31:16]=0 | tensor_dim1[15:0]
  g1[3] = CC << 16;          // tensor_dim1[31:16]=0 | tile_dim0
  g1[4] = 16;                // tile_dim1 = #valid indices; tile_dim2=0
  g1[5] = CC;                // tensor_dim0_stride[31:0]
  g1[6] = 0;                 // stride hi / dim1_stride (ignored in gather)
  g1[7] = 0;

  // ---- D# groups 2/3: sixteen 16-bit row indices
  i32x4 g2, g3;
#pragma unroll
  for (int k = 0; k < 4; ++k)
    g2[k] = (int)(ridx[2 * k] | (ridx[2 * k + 1] << 16));
#pragma unroll
  for (int k = 0; k < 4; ++k)
    g3[k] = (int)(ridx[8 + 2 * k] | (ridx[8 + 2 * k + 1] << 16));

  // clang-23 / therock toolchain 6-arg form:
  //   (u32x4 g0, i32x8 g1, i32x4 g2, i32x4 g3, i32x8 ext, i32 cpol)
  const i32x8 gx = {};
  __builtin_amdgcn_tensor_load_to_lds(g0, g1, g2, g3, gx, 0);
}
#endif

__global__ __launch_bounds__(256) void k_scores_argmax(
    const _Float16* __restrict__ m, const int* __restrict__ idx,
    int* __restrict__ dst_idx) {
  const int blk  = blockIdx.x;
  const int b    = blk / (RR / 128);
  const int mblk = blk % (RR / 128);
  const int* a_idx = idx + (size_t)b * NN;        // first RR entries
  const int* b_idx = a_idx + RR;                  // remaining NBT entries
  const int wave   = threadIdx.x >> 5;
  const int lane   = threadIdx.x & 31;
  const int lane16 = lane & 15;
  const int hl     = lane >> 4;                   // lane half (K-split per ISA)
  const int m0     = mblk * 128 + wave * 16;

  const _Float16* mb = m + (size_t)b * NN * CC;
  const int arow = a_idx[m0 + lane16];            // A fragment row for this lane
  const _Float16* aptr = mb + (size_t)arow * CC;

  __shared__ _Float16 Bt[2][16][CC];              // double buffer, LDS offset 0

  float best[8];
  int   bestc[8];
#pragma unroll
  for (int g = 0; g < 8; ++g) { best[g] = -3.0e38f; bestc[g] = 0; }

#if USE_TDM
  if (threadIdx.x < 32) tdm_issue_tile(mb, b_idx, 0);   // prologue: tile 0
#endif

  for (int j = 0; j < NBT / 16; ++j) {
    const int buf = j & 1;

#if USE_TDM
    if (threadIdx.x < 32) {   // one wave drives the DMA (TENSORcnt per-wave)
      if (j + 1 < NBT / 16) {
        tdm_issue_tile(mb, b_idx, j + 1);       // overlap next tile's DMA
        __builtin_amdgcn_s_wait_tensorcnt(1);   // in-order => tile j resident
      } else {
        __builtin_amdgcn_s_wait_tensorcnt(0);   // drain last tile
      }
    }
#else
    {
      // manual staging: 16 gathered b-rows x CC halves; 16B vector chunks
      const int r  = threadIdx.x >> 4;            // 0..15 : row of tile
      const int c0 = (threadIdx.x & 15) * 8;      // 8-half (16B) chunks
      const _Float16* brp = mb + (size_t)b_idx[j * 16 + r] * CC;
      for (int c = c0; c < CC; c += 16 * 8)
        *(uint4*)&Bt[buf][r][c] = *(const uint4*)&brp[c];
    }
#endif
    __syncthreads();   // publish tile j to all waves

    // Escape the LDS buffer so the optimizer knows the TDM wrote it.
    {
      _Float16* escp = &Bt[buf][0][0];
      asm volatile("" : "+v"(escp) : : "memory");
    }

    v8f acc = {};
#pragma unroll 4
    for (int kt = 0; kt < CC / 32; ++kt) {
      const int k0 = kt * 32;
      // A 16x32 f16 fragment: lanes 0-15 hold K {0-7,16-23}, lanes 16-31 {8-15,24-31}
      v16h afrag;
      {
        union { uint4 q[2]; v16h h; } u;
        u.q[0] = *(const uint4*)&aptr[k0 + hl * 8];
        u.q[1] = *(const uint4*)&aptr[k0 + 16 + hl * 8];
        afrag = u.h;
      }
      // B 32x16 f16 fragment: lane%16 = column (gathered b-row),
      // lanes 0-15 hold K {0-7,16-23}, lanes 16-31 {8-15,24-31}
      v16h bfrag;
      {
        union { uint4 q[2]; v16h h; } u;
        u.q[0] = *(const uint4*)&Bt[buf][lane16][k0 + hl * 8];
        u.q[1] = *(const uint4*)&Bt[buf][lane16][k0 + 16 + hl * 8];
        bfrag = u.h;
      }
      acc = __builtin_amdgcn_wmma_f32_16x16x32_f16(
          false, afrag, false, bfrag, (short)0, acc, false, false);
    }

    // C/D layout: VGPR g, lanes 0-15 -> (M=g, N=lane); lanes 16-31 -> (M=8+g)
    const int col = j * 16 + lane16;
#pragma unroll
    for (int g = 0; g < 8; ++g) {
      const float v = acc[g];
      if (v > best[g]) { best[g] = v; bestc[g] = col; }
    }

    __syncthreads();   // all waves done reading buf before it is overwritten
  }

  // argmax reduce across the 16 lanes of each half (same rows, distinct cols);
  // tie-break to the smaller column like jnp.argmax
#pragma unroll
  for (int off = 8; off > 0; off >>= 1) {
#pragma unroll
    for (int g = 0; g < 8; ++g) {
      const float ov = __shfl_xor(best[g], off, 16);
      const int   oc = __shfl_xor(bestc[g], off, 16);
      if (ov > best[g] || (ov == best[g] && oc < bestc[g])) {
        best[g] = ov; bestc[g] = oc;
      }
    }
  }
  if (lane16 == 0) {
#pragma unroll
    for (int g = 0; g < 8; ++g)
      dst_idx[(size_t)b * RR + m0 + g + 8 * hl] = bestc[g];
  }
}

// ---------------------------------------------------------------------------
// K6: merge. init out with dst rows + zero counts, atomic scatter of src rows,
// then divide by (1 + count).
// ---------------------------------------------------------------------------
__global__ __launch_bounds__(256) void k_init_out(const float* __restrict__ x,
                                                  const int* __restrict__ idx,
                                                  float* __restrict__ out,
                                                  float* __restrict__ cnt) {
  const int t = blockIdx.x;              // b*NBT + j
  const int b = t / NBT;
  const int j = t % NBT;
  const int brow = idx[(size_t)b * NN + RR + j];
  const float* src = x + ((size_t)b * NN + brow) * CC;
  float* o = out + (size_t)t * CC;
  for (int c = threadIdx.x; c < CC; c += 256) o[c] = src[c];
  if (threadIdx.x == 0) cnt[t] = 0.f;
}

__global__ __launch_bounds__(256) void k_scatter(const float* __restrict__ x,
                                                 const int* __restrict__ idx,
                                                 const int* __restrict__ dst_idx,
                                                 float* __restrict__ out,
                                                 float* __restrict__ cnt) {
  const int t = blockIdx.x;              // b*RR + r
  const int b = t / RR;
  const int r = t % RR;
  const int arow = idx[(size_t)b * NN + r];
  const int j = dst_idx[t];
  const float* src = x + ((size_t)b * NN + arow) * CC;
  float* o = out + ((size_t)b * NBT + j) * CC;
  for (int c = threadIdx.x; c < CC; c += 256) atomicAdd(&o[c], src[c]);
  if (threadIdx.x == 0) atomicAdd(&cnt[(size_t)b * NBT + j], 1.f);
}

__global__ __launch_bounds__(256) void k_div(float* __restrict__ out,
                                             const float* __restrict__ cnt) {
  const int t = blockIdx.x;              // b*NBT + j
  const float inv = 1.0f / (1.0f + cnt[t]);
  float* o = out + (size_t)t * CC;
  for (int c = threadIdx.x; c < CC; c += 256) o[c] *= inv;
}

// ---------------------------------------------------------------------------
extern "C" void kernel_launch(void* const* d_in, const int* in_sizes, int n_in,
                              void* d_out, int out_size, void* d_ws, size_t ws_size,
                              hipStream_t stream) {
  const float* x = (const float*)d_in[0];
  float* out = (float*)d_out;

  // workspace layout (~64.5 MB)
  char* ws = (char*)d_ws;
  _Float16* m   = (_Float16*)ws;                                  // BB*NN*CC*2
  float*    S   = (float*)(ws + (size_t)BB * NN * CC * 2);        // BB*CC
  float*    key = S + (size_t)BB * CC;                            // BB*NN
  int*      idx = (int*)(key + (size_t)BB * NN);                  // BB*NN
  int*      dsti = idx + (size_t)BB * NN;                         // BB*RR
  float*    cnt = (float*)(dsti + (size_t)BB * RR);               // BB*NBT

  k_normalize<<<BB * NN, 256, 0, stream>>>(x, m);
  k_colsum<<<(BB * CC) / 256, 256, 0, stream>>>(m, S);
  k_rowscore<<<(BB * NN) / 8, 256, 0, stream>>>(m, S, key);
  k_sort<<<BB, 1024, 0, stream>>>(key, idx);
  k_scores_argmax<<<BB * (RR / 128), 256, 0, stream>>>(m, idx, dsti);
  k_init_out<<<BB * NBT, 256, 0, stream>>>(x, idx, out, cnt);
  k_scatter<<<BB * RR, 256, 0, stream>>>(x, idx, dsti, out, cnt);
  k_div<<<BB * NBT, 256, 0, stream>>>(out, cnt);
}
